// GFGA_70239895158912
// MI455X (gfx1250) — matrix operand order repaired and checked by reading, not compiled, not run on hardware
//
#include <hip/hip_runtime.h>
#include <hip/hip_bf16.h>

// ---------------- model constants (match reference) ----------------
static constexpr int Bb  = 512;
static constexpr int NI  = 36;
static constexpr int NT  = 64;
static constexpr int DI  = 2048;
static constexpr int DT  = 768;
static constexpr int Gg  = 1024;
static constexpr int Hd  = 512;
static constexpr int Cc  = 32;
static constexpr int Dd  = 4;
static constexpr int Ll  = 2;
static constexpr int NHh = 8;
static constexpr int DHh = Hd / NHh;   // 64
static constexpr float EPSc = 0.1f;
static constexpr int ITERS  = 10;

// ---------------- WMMA types ----------------
typedef __attribute__((ext_vector_type(16))) __bf16 v16bf;
typedef __attribute__((ext_vector_type(8)))  float  v8f;

union FragBF { v16bf v; unsigned w[8]; __bf16 b[16]; };

__device__ inline int imin(int a, int b) { return a < b ? a : b; }

// native bf16 convert (RNE); packed per float pair to encourage
// v_cvt_pk_bf16_f32 selection
__device__ inline unsigned pkbf(float lo, float hi) {
  union { __bf16 b[2]; unsigned u; } t;
  t.b[0] = (__bf16)lo;
  t.b[1] = (__bf16)hi;
  return t.u;
}

__device__ inline void pack4(FragBF& f, int idx, float4 v) {   // idx multiple of 4
  f.w[(idx >> 1) + 0] = pkbf(v.x, v.y);
  f.w[(idx >> 1) + 1] = pkbf(v.z, v.w);
}

// A fragment per lane = 16 contiguous floats -> 4x b128 loads
__device__ inline void loadA(FragBF& f, const float* rowA, int k0, int hiA) {
  const float4* p0 = (const float4*)(rowA + k0 + hiA);
  const float4* p1 = (const float4*)(rowA + k0 + hiA + 8);
  pack4(f, 0, p0[0]); pack4(f, 4, p0[1]); pack4(f, 8, p1[0]); pack4(f, 12, p1[1]);
}
// B fragment per lane (row-major W) = 16 contiguous floats -> 4x b128 loads
__device__ inline void loadB(FragBF& f, const float* rowB, int k0, int hiB) {
  const float4* p = (const float4*)(rowB + k0 + hiB);
  pack4(f, 0, p[0]); pack4(f, 4, p[1]); pack4(f, 8, p[2]); pack4(f, 12, p[3]);
}
// B fragment, transposed W (stride ldw between consecutive K)
__device__ inline void loadBT(FragBF& f, const float* colB, int k0, int hiB, int ldw) {
  #pragma unroll
  for (int v = 0; v < 8; ++v) {
    int kb = k0 + hiB + 2 * v;
    f.w[v] = pkbf(colB[(long long)kb * ldw], colB[(long long)(kb + 1) * ldw]);
  }
}
// K-tail variants: clamped indices + select-to-zero (branch free)
__device__ inline void loadA_tail(FragBF& f, const float* rowA, int k0, int hiA, int K) {
  #pragma unroll
  for (int v = 0; v < 8; ++v) {
    int ka = k0 + hiA + 2 * v;
    float a0 = rowA[imin(ka,     K - 1)]; if (ka     >= K) a0 = 0.f;
    float a1 = rowA[imin(ka + 1, K - 1)]; if (ka + 1 >= K) a1 = 0.f;
    f.w[v] = pkbf(a0, a1);
  }
}
__device__ inline void loadB_tail(FragBF& f, const float* rowB, int k0, int hiB, int K,
                                  int ldw, int transW) {
  #pragma unroll
  for (int v = 0; v < 8; ++v) {
    int kb  = k0 + hiB + 2 * v;
    int kb0 = imin(kb, K - 1), kb1 = imin(kb + 1, K - 1);
    float b0 = transW ? rowB[(long long)kb0 * ldw] : rowB[kb0];
    float b1 = transW ? rowB[(long long)kb1 * ldw] : rowB[kb1];
    if (kb     >= K) b0 = 0.f;
    if (kb + 1 >= K) b1 = 0.f;
    f.w[v] = pkbf(b0, b1);
  }
}

__device__ inline float wave_sum(float v) {
  #pragma unroll
  for (int m = 16; m >= 1; m >>= 1) v += __shfl_xor(v, m, 32);
  return v;
}
__device__ inline float wave_max(float v) {
  #pragma unroll
  for (int m = 16; m >= 1; m >>= 1) v = fmaxf(v, __shfl_xor(v, m, 32));
  return v;
}

// =====================================================================
// Generic batched GEMM:  Y = act(alpha * X @ W^T + bias) (+ resid)
// batch z: b = z/nh, h = z%nh.
// Wave computes a 2x2 grid of 16x16 tiles (32M x 32N): 2 A-frags + 2 B-frags
// feed 4 WMMAs per K-step -> 2x operand reuse vs 1 tile/wave.
// M/N edges: source row/col clamped (garbage lands only in discarded D
// rows/cols). K tail: clamp + select-to-zero. No exec-mask branching.
// =====================================================================
__global__ __launch_bounds__(128) void k_linear(
    const float* __restrict__ X, long long xb, long long xh, int ldx,
    const float* __restrict__ W, long long wbs, long long wh, int ldw, int transW,
    const float* __restrict__ bias,
    const float* __restrict__ R, long long rb, long long rh, int ldr,
    float* __restrict__ Y, long long yb, long long yh, int ldy,
    int M, int N, int K, int nh, float alpha, int act)
{
  int z = blockIdx.z;
  int b = z / nh, h = z - b * nh;
  const float* Xp = X + (long long)b * xb + (long long)h * xh;
  const float* Wp = W + (long long)b * wbs + (long long)h * wh;
  float*       Yp = Y + (long long)b * yb + (long long)h * yh;
  const float* Rp = R ? (R + (long long)b * rb + (long long)h * rh) : nullptr;

  const int lane = threadIdx.x & 31;
  const int wave = threadIdx.x >> 5;
  const int m0 = blockIdx.x * 32;                  // two M tiles: m0, m0+16
  const int n0 = blockIdx.y * 128 + wave * 32;     // two N tiles: n0, n0+16
  if (n0 >= N) return;                             // uniform per wave

  const int l15 = lane & 15;
  const int hiA = (lane >> 4) * 16;
  const int hiB = (lane >> 4) * 16;

  const int mA0 = imin(m0 + l15,      M - 1);
  const int mA1 = imin(m0 + 16 + l15, M - 1);
  const int nB0 = imin(n0 + l15,      N - 1);
  const int nB1 = imin(n0 + 16 + l15, N - 1);

  const float* rowA0 = Xp + (long long)mA0 * ldx;
  const float* rowA1 = Xp + (long long)mA1 * ldx;
  const float* rowB0 = transW ? (Wp + nB0) : (Wp + (long long)nB0 * ldw);
  const float* rowB1 = transW ? (Wp + nB1) : (Wp + (long long)nB1 * ldw);

  v8f a00 = {0.f,0.f,0.f,0.f,0.f,0.f,0.f,0.f};
  v8f a01 = a00, a10 = a00, a11 = a00;

  int k0 = 0;
  if (!transW) {
    for (; k0 + 32 <= K; k0 += 32) {
      FragBF fa0, fa1, fb0, fb1;
      loadA(fa0, rowA0, k0, hiA);
      loadA(fa1, rowA1, k0, hiA);
      loadB(fb0, rowB0, k0, hiB);
      loadB(fb1, rowB1, k0, hiB);
      a00 = __builtin_amdgcn_wmma_f32_16x16x32_bf16(false, fa0.v, false, fb0.v, (short)0, a00, false, false);
      a01 = __builtin_amdgcn_wmma_f32_16x16x32_bf16(false, fa0.v, false, fb1.v, (short)0, a01, false, false);
      a10 = __builtin_amdgcn_wmma_f32_16x16x32_bf16(false, fa1.v, false, fb0.v, (short)0, a10, false, false);
      a11 = __builtin_amdgcn_wmma_f32_16x16x32_bf16(false, fa1.v, false, fb1.v, (short)0, a11, false, false);
    }
  } else {
    for (; k0 + 32 <= K; k0 += 32) {
      FragBF fa0, fa1, fb0, fb1;
      loadA(fa0, rowA0, k0, hiA);
      loadA(fa1, rowA1, k0, hiA);
      loadBT(fb0, rowB0, k0, hiB, ldw);
      loadBT(fb1, rowB1, k0, hiB, ldw);
      a00 = __builtin_amdgcn_wmma_f32_16x16x32_bf16(false, fa0.v, false, fb0.v, (short)0, a00, false, false);
      a01 = __builtin_amdgcn_wmma_f32_16x16x32_bf16(false, fa0.v, false, fb1.v, (short)0, a01, false, false);
      a10 = __builtin_amdgcn_wmma_f32_16x16x32_bf16(false, fa1.v, false, fb0.v, (short)0, a10, false, false);
      a11 = __builtin_amdgcn_wmma_f32_16x16x32_bf16(false, fa1.v, false, fb1.v, (short)0, a11, false, false);
    }
  }
  if (k0 < K) {   // K tail (only the attention A*V GEMM, K=36)
    FragBF fa0, fa1, fb0, fb1;
    loadA_tail(fa0, rowA0, k0, hiA, K);
    loadA_tail(fa1, rowA1, k0, hiA, K);
    loadB_tail(fb0, rowB0, k0, hiB, K, ldw, transW);
    loadB_tail(fb1, rowB1, k0, hiB, K, ldw, transW);
    a00 = __builtin_amdgcn_wmma_f32_16x16x32_bf16(false, fa0.v, false, fb0.v, (short)0, a00, false, false);
    a01 = __builtin_amdgcn_wmma_f32_16x16x32_bf16(false, fa0.v, false, fb1.v, (short)0, a01, false, false);
    a10 = __builtin_amdgcn_wmma_f32_16x16x32_bf16(false, fa1.v, false, fb0.v, (short)0, a10, false, false);
    a11 = __builtin_amdgcn_wmma_f32_16x16x32_bf16(false, fa1.v, false, fb1.v, (short)0, a11, false, false);
  }

  // epilogue: 2x2 tiles, store-guarded at true M/N edges
  #pragma unroll
  for (int mt = 0; mt < 2; ++mt) {
    #pragma unroll
    for (int nt = 0; nt < 2; ++nt) {
      v8f acc = (mt == 0) ? (nt == 0 ? a00 : a01) : (nt == 0 ? a10 : a11);
      int nOut = n0 + nt * 16 + l15;
      if (nOut >= N) continue;
      float bi = bias ? bias[nOut] : 0.f;
      #pragma unroll
      for (int r = 0; r < 8; ++r) {
        int m = m0 + mt * 16 + r + (lane >> 4) * 8;
        if (m >= M) continue;
        float y = alpha * acc[r] + bi;
        if (act == 1) y = 0.5f * y * (1.f + erff(y * 0.70710678118654752440f)); // exact GELU
        if (Rp) y += Rp[(long long)m * ldr + nOut];
        Yp[(long long)m * ldy + nOut] = y;
      }
    }
  }
}

// ---------------- LayerNorm over last dim (one wave per row) ----------------
__global__ __launch_bounds__(256) void k_ln(const float* __restrict__ x,
    const float* __restrict__ g, const float* __restrict__ bta,
    float* __restrict__ y, int rows, int H)
{
  int row = blockIdx.x * 8 + (threadIdx.x >> 5);
  if (row >= rows) return;
  int lane = threadIdx.x & 31;
  const float* xr = x + (long long)row * H;
  float s = 0.f, s2 = 0.f;
  for (int i = lane; i < H; i += 32) { float v = xr[i]; s += v; s2 += v * v; }
  s = wave_sum(s); s2 = wave_sum(s2);
  float m   = s / (float)H;
  float var = s2 / (float)H - m * m;
  float inv = rsqrtf(var + 1e-5f);
  float* yr = y + (long long)row * H;
  for (int i = lane; i < H; i += 32) yr[i] = (xr[i] - m) * inv * g[i] + bta[i];
}

// ---------------- L2 normalize rows (max(norm,1e-12)) ----------------
__global__ __launch_bounds__(256) void k_l2norm(float* __restrict__ x, int rows, int H)
{
  int row = blockIdx.x * 8 + (threadIdx.x >> 5);
  if (row >= rows) return;
  int lane = threadIdx.x & 31;
  float* xr = x + (long long)row * H;
  float s2 = 0.f;
  for (int i = lane; i < H; i += 32) { float v = xr[i]; s2 += v * v; }
  s2 = wave_sum(s2);
  float inv = 1.f / fmaxf(sqrtf(s2), 1e-12f);
  for (int i = lane; i < H; i += 32) xr[i] *= inv;
}

// ---------------- row softmax with optional key-padding mask ----------------
__global__ __launch_bounds__(256) void k_softmax(float* __restrict__ x, int rows, int S,
    const int* __restrict__ lengths, int rowsPerBatch)
{
  int row = blockIdx.x * 8 + (threadIdx.x >> 5);
  if (row >= rows) return;
  int lane = threadIdx.x & 31;
  float* xr = x + (long long)row * S;
  int len = lengths ? lengths[row / rowsPerBatch] : S;
  float mx = -1e30f;
  for (int i = lane; i < S; i += 32) if (i < len) mx = fmaxf(mx, xr[i]);
  mx = wave_max(mx);
  float s = 0.f;
  for (int i = lane; i < S; i += 32) if (i < len) s += __expf(xr[i] - mx);
  s = wave_sum(s);
  float inv = 1.f / s;
  for (int i = lane; i < S; i += 32)
    xr[i] = (i < len) ? __expf(xr[i] - mx) * inv : 0.f;
}

// ---------------- out[b,n] = alpha * dot(q[b,:], p[b,n,:]) ----------------
__global__ __launch_bounds__(256) void k_rowdot(const float* __restrict__ q,
    const float* __restrict__ p, float* __restrict__ out, int Bn, int N, int H, float alpha)
{
  int row = blockIdx.x * 8 + (threadIdx.x >> 5);
  if (row >= Bn) return;
  int lane = threadIdx.x & 31;
  int b = row / N;
  const float* qr = q + (long long)b * H;
  const float* pr = p + (long long)row * H;
  float s = 0.f;
  for (int i = lane; i < H; i += 32) s += qr[i] * pr[i];
  s = wave_sum(s);
  if (lane == 0) out[row] = s * alpha;
}

// ---------------- broadcast [n] -> [B, n] ----------------
__global__ void k_bcast(const float* __restrict__ src, float* __restrict__ y, int n, int B)
{
  int i = blockIdx.x * 256 + threadIdx.x;
  if (i >= n) return;
  float v = src[i];
  for (int b = 0; b < B; ++b) y[(long long)b * n + i] = v;
}

// ---------------- concat [B,Ma,H] + shared [Md,H] -> [B,Ma+Md,H] ----------------
__global__ void k_concat(const float* __restrict__ a, const float* __restrict__ shr,
                         float* __restrict__ y, int Ma, int Md, int H)
{
  int row = blockIdx.x;
  int tot = Ma + Md;
  int b = row / tot, r = row - b * tot;
  const float* src = (r < Ma) ? (a + ((long long)b * Ma + r) * H)
                              : (shr + (long long)(r - Ma) * H);
  float* dst = y + (long long)row * H;
  for (int i = threadIdx.x; i < H; i += blockDim.x) dst[i] = src[i];
}

// ---------------- per-batch Sinkhorn + final contraction ----------------
static constexpr int RI = NI + Dd;   // 40
static constexpr int RJ = NT + Dd;   // 68
__global__ __launch_bounds__(128) void k_sinkhorn(const float* __restrict__ sim,
    const float* __restrict__ img_s, const float* __restrict__ txt_s,
    float* __restrict__ out)
{
  __shared__ float S[RI * RJ], Km[RI * RJ], u[RI], v[RJ], mu[RI], mv[RJ], red[128];
  int b = blockIdx.x, tid = threadIdx.x;
  const float* sb = sim + (long long)b * RI * RJ;
  for (int i = tid; i < RI * RJ; i += 128) {
    float s = sb[i];
    S[i]  = s;
    Km[i] = __expf(-(1.f - s) / EPSc);
  }
  if (tid < RI) { mu[tid] = (tid < NI) ? img_s[(long long)b * NI + tid] : 1.f / Dd; u[tid] = 1.f; }
  if (tid < RJ) { mv[tid] = (tid < NT) ? txt_s[(long long)b * NT + tid] : 1.f / Dd; v[tid] = 1.f; }
  __syncthreads();
  for (int it = 0; it < ITERS; ++it) {
    if (tid < RI) {
      float s = 0.f;
      for (int j = 0; j < RJ; ++j) s += Km[tid * RJ + j] * v[j];
      u[tid] = mu[tid] / s;
    }
    __syncthreads();
    if (tid < RJ) {
      float s = 0.f;
      for (int i = 0; i < RI; ++i) s += Km[i * RJ + tid] * u[i];
      v[tid] = mv[tid] / s;
    }
    __syncthreads();
  }
  float acc = 0.f;
  for (int idx = tid; idx < NI * NT; idx += 128) {
    int i = idx / NT, j = idx - i * NT;
    acc += u[i] * Km[i * RJ + j] * v[j] * S[i * RJ + j];
  }
  red[tid] = acc;
  __syncthreads();
  for (int st = 64; st >= 1; st >>= 1) {
    if (tid < st) red[tid] += red[tid + st];
    __syncthreads();
  }
  if (tid == 0) out[b] = red[0];
}

// =====================================================================
// Host orchestration
// =====================================================================
extern "C" void kernel_launch(void* const* d_in, const int* in_sizes, int n_in,
                              void* d_out, int out_size, void* d_ws, size_t ws_size,
                              hipStream_t stream) {
  (void)in_sizes; (void)n_in; (void)out_size; (void)ws_size;
  // inputs in setup_inputs() dict order
  const float* img_global = (const float*)d_in[0];
  const float* img_nodes  = (const float*)d_in[1];
  const float* txt_global = (const float*)d_in[2];
  const float* txt_nodes  = (const float*)d_in[3];
  const int*   lengths    = (const int*)  d_in[4];
  const float* img_in_w   = (const float*)d_in[5];
  const float* img_in_b   = (const float*)d_in[6];
  const float* txt_in_w   = (const float*)d_in[7];
  const float* txt_in_b   = (const float*)d_in[8];
  const float* img_conc   = (const float*)d_in[9];
  const float* txt_conc   = (const float*)d_in[10];
  const float* img_dust   = (const float*)d_in[11];
  const float* txt_dust   = (const float*)d_in[12];
  const float* ln_g       = (const float*)d_in[13];
  const float* ln_b       = (const float*)d_in[14];
  const float* attn_in_w  = (const float*)d_in[15];
  const float* attn_in_b  = (const float*)d_in[16];
  const float* attn_out_w = (const float*)d_in[17];
  const float* attn_out_b = (const float*)d_in[18];
  const float* ff1_w      = (const float*)d_in[19];
  const float* ff1_b      = (const float*)d_in[20];
  const float* ff2_w      = (const float*)d_in[21];
  const float* ff2_b      = (const float*)d_in[22];
  const float* lln_g      = (const float*)d_in[23];
  const float* lln_b      = (const float*)d_in[24];
  const float* db_w       = (const float*)d_in[25];
  const float* db_b       = (const float*)d_in[26];
  const float* proj_w     = (const float*)d_in[27];
  const float* proj_b     = (const float*)d_in[28];
  const float* qg_w       = (const float*)d_in[29];
  const float* qg_b       = (const float*)d_in[30];
  float* out = (float*)d_out;

  // ---- workspace carve-up (floats) ----
  float* base = (float*)d_ws;
  size_t off = 0;
  auto alloc = [&](size_t n) { float* r = base + off; off += n; return r; };
  float* img_n   = alloc((size_t)Bb * NI * Hd);
  float* txt_n   = alloc((size_t)Bb * NT * Hd);
  float* img_c   = alloc((size_t)Bb * Cc * Hd);
  float* txt_c   = alloc((size_t)Bb * Cc * Hd);
  float* img_c2  = alloc((size_t)Bb * (Cc + Dd) * Hd);
  float* txt_c2  = alloc((size_t)Bb * (Cc + Dd) * Hd);
  float* ccp     = alloc((size_t)Cc * Hd);
  float* cdb_i   = alloc((size_t)Dd * Hd);
  float* cdb_t   = alloc((size_t)Dd * Hd);
  float* dbp_i   = alloc((size_t)Dd * Hd);
  float* dbp_t   = alloc((size_t)Dd * Hd);
  float* sln     = alloc((size_t)Bb * NT * Hd);
  float* tln     = alloc((size_t)Bb * NT * Hd);
  float* qb      = alloc((size_t)Bb * NT * Hd);
  float* kb      = alloc((size_t)Bb * NT * Hd);
  float* vb      = alloc((size_t)Bb * NT * Hd);
  float* ao      = alloc((size_t)Bb * NT * Hd);
  float* att     = alloc((size_t)Bb * NHh * NT * NT);
  float* ffh     = alloc((size_t)Bb * NT * 4 * Hd);
  float* qg_i    = alloc((size_t)Bb * Hd);
  float* qg_t    = alloc((size_t)Bb * Hd);
  float* img_s   = alloc((size_t)Bb * NI);
  float* txt_s   = alloc((size_t)Bb * NT);
  float* img_all = alloc((size_t)Bb * RI * Hd);
  float* txt_all = alloc((size_t)Bb * RJ * Hd);
  float* simb    = alloc((size_t)Bb * RI * RJ);

  // ---- launch helpers ----
  auto lin = [&](const float* X, long long xb2, long long xh, int ldx,
                 const float* W, long long wbs, long long wh, int ldw, int transW,
                 const float* bias, const float* R, long long rb, long long rh, int ldr,
                 float* Y, long long yb2, long long yh, int ldy,
                 int M, int N, int K, int nh, int batches, float alpha, int act) {
    dim3 g((M + 31) / 32, (N + 127) / 128, batches * nh);
    k_linear<<<g, dim3(128), 0, stream>>>(X, xb2, xh, ldx, W, wbs, wh, ldw, transW,
                                          bias, R, rb, rh, ldr, Y, yb2, yh, ldy,
                                          M, N, K, nh, alpha, act);
  };
  auto ln = [&](const float* x, const float* g, const float* b2, float* y, int rows) {
    k_ln<<<(rows + 7) / 8, 256, 0, stream>>>(x, g, b2, y, rows, Hd);
  };
  auto smax = [&](float* x, int rows, int S, const int* lens, int rpb) {
    k_softmax<<<(rows + 7) / 8, 256, 0, stream>>>(x, rows, S, lens, rpb);
  };

  // ---- input projections + LN ----
  lin(img_nodes, (long long)NI * DI, 0, DI, img_in_w, 0, 0, DI, 0, img_in_b,
      nullptr, 0, 0, 0, tln, (long long)NI * Hd, 0, Hd, NI, Hd, DI, 1, Bb, 1.f, 0);
  ln(tln, ln_g + 0 * Hd, ln_b + 0 * Hd, img_n, Bb * NI);
  lin(txt_nodes, (long long)NT * DT, 0, DT, txt_in_w, 0, 0, DT, 0, txt_in_b,
      nullptr, 0, 0, 0, tln, (long long)NT * Hd, 0, Hd, NT, Hd, DT, 1, Bb, 1.f, 0);
  ln(tln, ln_g + 1 * Hd, ln_b + 1 * Hd, txt_n, Bb * NT);

  // concepts / dustbins (batch-independent LN, then broadcast concepts)
  ln(img_conc, ln_g + 4 * Hd, ln_b + 4 * Hd, ccp, Cc);
  k_bcast<<<(Cc * Hd + 255) / 256, 256, 0, stream>>>(ccp, img_c, Cc * Hd, Bb);
  ln(txt_conc, ln_g + 5 * Hd, ln_b + 5 * Hd, ccp, Cc);
  k_bcast<<<(Cc * Hd + 255) / 256, 256, 0, stream>>>(ccp, txt_c, Cc * Hd, Bb);
  ln(img_dust, ln_g + 6 * Hd, ln_b + 6 * Hd, cdb_i, Dd);
  ln(txt_dust, ln_g + 7 * Hd, ln_b + 7 * Hd, cdb_t, Dd);

  // ---- one cross-attention + FFN layer ----
  auto xlayer = [&](int s, int l, const float* src, int S, float* tgt, int T,
                    const int* lens) {
    const int sl = s * Ll + l;
    const float* aw  = attn_in_w  + (size_t)sl * 3 * Hd * Hd;
    const float* abv = attn_in_b  + (size_t)sl * 3 * Hd;
    const float* ow  = attn_out_w + (size_t)sl * Hd * Hd;
    const float* obv = attn_out_b + (size_t)sl * Hd;
    const float* f1w = ff1_w + (size_t)sl * 4 * Hd * Hd;
    const float* f1b = ff1_b + (size_t)sl * 4 * Hd;
    const float* f2w = ff2_w + (size_t)sl * Hd * 4 * Hd;
    const float* f2b = ff2_b + (size_t)sl * Hd;
    const float* lg  = lln_g + (size_t)sl * 3 * Hd;
    const float* lb2 = lln_b + (size_t)sl * 3 * Hd;

    ln(src, lg + 0 * Hd, lb2 + 0 * Hd, sln, Bb * S);
    ln(tgt, lg + 1 * Hd, lb2 + 1 * Hd, tln, Bb * T);
    // q/k/v projections
    lin(tln, (long long)T * Hd, 0, Hd, aw, 0, 0, Hd, 0, abv,
        nullptr, 0, 0, 0, qb, (long long)T * Hd, 0, Hd, T, Hd, Hd, 1, Bb, 1.f, 0);
    lin(sln, (long long)S * Hd, 0, Hd, aw + (size_t)Hd * Hd, 0, 0, Hd, 0, abv + Hd,
        nullptr, 0, 0, 0, kb, (long long)S * Hd, 0, Hd, S, Hd, Hd, 1, Bb, 1.f, 0);
    lin(sln, (long long)S * Hd, 0, Hd, aw + (size_t)2 * Hd * Hd, 0, 0, Hd, 0, abv + 2 * Hd,
        nullptr, 0, 0, 0, vb, (long long)S * Hd, 0, Hd, S, Hd, Hd, 1, Bb, 1.f, 0);
    // scores[b,h,t,s] = q . k / sqrt(DH)
    lin(qb, (long long)T * Hd, DHh, Hd, kb, (long long)S * Hd, DHh, Hd, 0,
        nullptr, nullptr, 0, 0, 0,
        att, (long long)NHh * T * S, (long long)T * S, S, T, S, DHh, NHh, Bb, 0.125f, 0);
    smax(att, Bb * NHh * T, S, lens, NHh * T);
    // o = probs @ V  (W = V with transposed access, K-dim = S)
    lin(att, (long long)NHh * T * S, (long long)T * S, S,
        vb, (long long)S * Hd, DHh, Hd, 1,
        nullptr, nullptr, 0, 0, 0,
        ao, (long long)T * Hd, DHh, Hd, T, DHh, S, NHh, Bb, 1.f, 0);
    // out-proj + residual into tgt
    lin(ao, (long long)T * Hd, 0, Hd, ow, 0, 0, Hd, 0, obv,
        tgt, (long long)T * Hd, 0, Hd,
        tgt, (long long)T * Hd, 0, Hd, T, Hd, Hd, 1, Bb, 1.f, 0);
    // FFN
    ln(tgt, lg + 2 * Hd, lb2 + 2 * Hd, tln, Bb * T);
    lin(tln, (long long)T * Hd, 0, Hd, f1w, 0, 0, Hd, 0, f1b,
        nullptr, 0, 0, 0, ffh, (long long)T * 4 * Hd, 0, 4 * Hd, T, 4 * Hd, Hd, 1, Bb, 1.f, 1);
    lin(ffh, (long long)T * 4 * Hd, 0, 4 * Hd, f2w, 0, 0, 4 * Hd, 0, f2b,
        tgt, (long long)T * Hd, 0, Hd,
        tgt, (long long)T * Hd, 0, Hd, T, Hd, 4 * Hd, 1, Bb, 1.f, 0);
  };

  // ---- layer loop ----
  for (int l = 0; l < Ll; ++l) {
    xlayer(0, l, img_n, NI, img_c, Cc, nullptr);
    xlayer(1, l, txt_n, NT, txt_c, Cc, lengths);
    // projected dustbins (batch-independent)
    lin(cdb_i, 0, 0, Hd, db_w + (size_t)(0 * Ll + l) * Hd * Hd, 0, 0, Hd, 0,
        db_b + (size_t)(0 * Ll + l) * Hd, nullptr, 0, 0, 0,
        dbp_i, 0, 0, Hd, Dd, Hd, Hd, 1, 1, 1.f, 0);
    lin(cdb_t, 0, 0, Hd, db_w + (size_t)(1 * Ll + l) * Hd * Hd, 0, 0, Hd, 0,
        db_b + (size_t)(1 * Ll + l) * Hd, nullptr, 0, 0, 0,
        dbp_t, 0, 0, Hd, Dd, Hd, Hd, 1, 1, 1.f, 0);
    k_concat<<<Bb * (Cc + Dd), 256, 0, stream>>>(img_c, dbp_i, img_c2, Cc, Dd, Hd);
    k_concat<<<Bb * (Cc + Dd), 256, 0, stream>>>(txt_c, dbp_t, txt_c2, Cc, Dd, Hd);
    xlayer(2, l, img_c2, Cc + Dd, txt_n, NT, nullptr);
    xlayer(3, l, txt_c2, Cc + Dd, img_n, NI, nullptr);
  }

  // ---- final projections ----
  ln(img_n, ln_g + 2 * Hd, ln_b + 2 * Hd, sln, Bb * NI);
  lin(sln, (long long)NI * Hd, 0, Hd, proj_w + (size_t)0 * Hd * Hd, 0, 0, Hd, 0,
      proj_b + 0 * Hd, nullptr, 0, 0, 0,
      img_n, (long long)NI * Hd, 0, Hd, NI, Hd, Hd, 1, Bb, 1.f, 0);
  ln(txt_n, ln_g + 3 * Hd, ln_b + 3 * Hd, sln, Bb * NT);
  lin(sln, (long long)NT * Hd, 0, Hd, proj_w + (size_t)1 * Hd * Hd, 0, 0, Hd, 0,
      proj_b + 1 * Hd, nullptr, 0, 0, 0,
      txt_n, (long long)NT * Hd, 0, Hd, NT, Hd, Hd, 1, Bb, 1.f, 0);

  // global queries: [B,G] @ qg_w^T + qg_b
  lin(img_global, 0, 0, Gg, qg_w + (size_t)0 * Hd * Gg, 0, 0, Gg, 0, qg_b + 0 * Hd,
      nullptr, 0, 0, 0, qg_i, 0, 0, Hd, Bb, Hd, Gg, 1, 1, 1.f, 0);
  lin(txt_global, 0, 0, Gg, qg_w + (size_t)1 * Hd * Gg, 0, 0, Gg, 0, qg_b + 1 * Hd,
      nullptr, 0, 0, 0, qg_t, 0, 0, Hd, Bb, Hd, Gg, 1, 1, 1.f, 0);

  // scoring projections + dot + softmax
  const float invSqrtH = 0.04419417382415922f;   // 1/sqrt(512)
  lin(img_n, (long long)NI * Hd, 0, Hd, proj_w + (size_t)2 * Hd * Hd, 0, 0, Hd, 0,
      proj_b + 2 * Hd, nullptr, 0, 0, 0,
      qb, (long long)NI * Hd, 0, Hd, NI, Hd, Hd, 1, Bb, 1.f, 0);
  k_rowdot<<<(Bb * NI + 7) / 8, 256, 0, stream>>>(qg_i, qb, img_s, Bb * NI, NI, Hd, invSqrtH);
  smax(img_s, Bb, NI, nullptr, 1);
  lin(txt_n, (long long)NT * Hd, 0, Hd, proj_w + (size_t)3 * Hd * Hd, 0, 0, Hd, 0,
      proj_b + 3 * Hd, nullptr, 0, 0, 0,
      kb, (long long)NT * Hd, 0, Hd, NT, Hd, Hd, 1, Bb, 1.f, 0);
  k_rowdot<<<(Bb * NT + 7) / 8, 256, 0, stream>>>(qg_t, kb, txt_s, Bb * NT, NT, Hd, invSqrtH);
  smax(txt_s, Bb, NT, lengths, 1);

  // dustbin projections for the "all" tensors (batch-independent)
  lin(cdb_i, 0, 0, Hd, proj_w + (size_t)4 * Hd * Hd, 0, 0, Hd, 0, proj_b + 4 * Hd,
      nullptr, 0, 0, 0, dbp_i, 0, 0, Hd, Dd, Hd, Hd, 1, 1, 1.f, 0);
  lin(cdb_t, 0, 0, Hd, proj_w + (size_t)5 * Hd * Hd, 0, 0, Hd, 0, proj_b + 5 * Hd,
      nullptr, 0, 0, 0, dbp_t, 0, 0, Hd, Dd, Hd, Hd, 1, 1, 1.f, 0);
  k_concat<<<Bb * RI, 256, 0, stream>>>(img_n, dbp_i, img_all, NI, Dd, Hd);
  k_concat<<<Bb * RJ, 256, 0, stream>>>(txt_n, dbp_t, txt_all, NT, Dd, Hd);
  k_l2norm<<<(Bb * RI + 7) / 8, 256, 0, stream>>>(img_all, Bb * RI, Hd);
  k_l2norm<<<(Bb * RJ + 7) / 8, 256, 0, stream>>>(txt_all, Bb * RJ, Hd);

  // sim[b] = img_nrm[b] @ txt_nrm[b]^T   (batched W, WMMA with M/N edge guards)
  lin(img_all, (long long)RI * Hd, 0, Hd,
      txt_all, (long long)RJ * Hd, 0, Hd, 0,
      nullptr, nullptr, 0, 0, 0,
      simb, (long long)RI * RJ, 0, RJ, RI, RJ, Hd, 1, Bb, 1.f, 0);

  // Sinkhorn + final contraction -> out[B]
  k_sinkhorn<<<Bb, 128, 0, stream>>>(simb, img_s, txt_s, out);
}
